// GraphSageLayer_58609123721516
// MI455X (gfx1250) — compile-verified, matrix operand order checked by reading, not run on hardware
//
#include <hip/hip_runtime.h>
#include <stdint.h>

#define NN 40000
#define NE 640000
#define DD 128
#define SWS 136                 // padded LDS row stride (halfwords): kills bank conflicts
#define BN_EPS 1e-5f

typedef __attribute__((ext_vector_type(16))) __bf16 v16bf;
typedef __attribute__((ext_vector_type(8)))  __bf16 v8bf;
typedef __attribute__((ext_vector_type(8)))  float  v8f;

// Load 8 contiguous f32 (32B aligned), scale, convert to 8 packed bf16
// using the native gfx1250 f32->bf16 conversion.
__device__ __forceinline__ v8bf cvt8(const float* __restrict__ p, float scale) {
  float4 a = ((const float4*)p)[0];
  float4 b = ((const float4*)p)[1];
  v8bf r;
  r[0] = (__bf16)(a.x * scale); r[1] = (__bf16)(a.y * scale);
  r[2] = (__bf16)(a.z * scale); r[3] = (__bf16)(a.w * scale);
  r[4] = (__bf16)(b.x * scale); r[5] = (__bf16)(b.y * scale);
  r[6] = (__bf16)(b.z * scale); r[7] = (__bf16)(b.w * scale);
  return r;
}

#define CAT16 0,1,2,3,4,5,6,7,8,9,10,11,12,13,14,15

// ---------------------------------------------------------------- kernel 1
__global__ void zero_kernel(float* __restrict__ p, int n) {
  int i = blockIdx.x * blockDim.x + threadIdx.x;
  if (i < n) p[i] = 0.0f;
}

// ---------------------------------------------------------------- kernel 2
// One wave (32 lanes) per edge: lane i moves features [4i, 4i+4).
__global__ __launch_bounds__(256) void edge_kernel(
    const float* __restrict__ h, const int* __restrict__ src,
    const int* __restrict__ dst, float* __restrict__ agg,
    float* __restrict__ deg) {
  const long long gid = (long long)blockIdx.x * blockDim.x + threadIdx.x;
  const int e = (int)(gid >> 5);
  const int lane = (int)(gid & 31);
  if (e >= NE) return;
  const int s = src[e];
  const int d = dst[e];
  float4 v = ((const float4*)(h + (size_t)s * DD))[lane];
  float* ap = agg + (size_t)d * DD + lane * 4;
  atomicAdd(ap + 0, v.x);
  atomicAdd(ap + 1, v.y);
  atomicAdd(ap + 2, v.z);
  atomicAdd(ap + 3, v.w);
  if (lane == 0) atomicAdd(deg + d, 1.0f);
}

// ---------------------------------------------------------------- kernel 3
// Fused: out_pre = relu(h@Wself + (agg/deg)@Wneigh + b) * snorm
// Also accumulates per-feature sum / sumsq for BN into gsum / gsq.
// NN % 16 == 0 -> each wave's 16-row strip is fully valid or fully invalid,
// so all row guards are wave-uniform (EXEC stays all-ones for WMMA).
__global__ __launch_bounds__(256) void sage_gemm_kernel(
    const float* __restrict__ h, const float* __restrict__ snorm,
    const float* __restrict__ Wself, const float* __restrict__ Wneigh,
    const float* __restrict__ bias, const float* __restrict__ agg,
    const float* __restrict__ deg, float* __restrict__ out,
    float* __restrict__ gsum, float* __restrict__ gsq) {
  // W staged transposed (Wt[n][k]) as bf16, padded row stride SWS so that
  // consecutive columns (lanes) land on different LDS banks.
  __shared__ __align__(16) __bf16 sW[2][DD * SWS];
  __shared__ float ssum[DD];
  __shared__ float ssq[DD];

  const int t = threadIdx.x;
  for (int i = t; i < DD * DD; i += 256) {
    const int n = i >> 7;
    const int k = i & (DD - 1);
    sW[0][n * SWS + k] = (__bf16)Wself[k * DD + n];
    sW[1][n * SWS + k] = (__bf16)Wneigh[k * DD + n];
  }
  if (t < DD) { ssum[t] = 0.0f; ssq[t] = 0.0f; }
  __syncthreads();

  const int wave = t >> 5;                 // 8 waves, one 16-row strip each
  const int lane = t & 31;
  const int rowbase = blockIdx.x * 128 + wave * 16;

  if (rowbase < NN) {                      // wave-uniform validity of strip
    const int arow = rowbase + (lane & 15);     // A-fragment row (valid)
    const int klo  = (lane < 16) ? 0 : 8;       // 16-bit A/B lane K-split
    const int ncol  = lane & 15;
    const int mbase = (lane < 16) ? 0 : 8;      // C/D layout row split

    const float invdeg = __builtin_amdgcn_rcpf(fmaxf(deg[arow], 1.0f));

    // A-fragments for all four K=32 steps (reused across all 8 col tiles).
    v16bf Ah[4], An[4];
#pragma unroll
    for (int kk = 0; kk < 4; ++kk) {
      const float* hp = h   + (size_t)arow * DD + kk * 32 + klo;
      const float* ap = agg + (size_t)arow * DD + kk * 32 + klo;
      Ah[kk] = __builtin_shufflevector(cvt8(hp, 1.0f), cvt8(hp + 16, 1.0f),
                                       CAT16);
      An[kk] = __builtin_shufflevector(cvt8(ap, invdeg), cvt8(ap + 16, invdeg),
                                       CAT16);
    }

    // Hoisted epilogue operands (independent of column tile).
    float sn[8];
    float bv[8];
#pragma unroll
    for (int v = 0; v < 8; ++v) sn[v] = snorm[rowbase + mbase + v];
#pragma unroll
    for (int ct = 0; ct < 8; ++ct) bv[ct] = bias[ct * 16 + ncol];

    for (int ct = 0; ct < 8; ++ct) {
      const int c = ct * 16 + ncol;

      // Batch all 16 B-fragment LDS loads for this tile first, then WMMA.
      v16bf Bs[4], Bn[4];
#pragma unroll
      for (int kk = 0; kk < 4; ++kk) {
        const __bf16* bsp = &sW[0][c * SWS + kk * 32 + klo];
        Bs[kk] = __builtin_shufflevector(*(const v8bf*)bsp,
                                         *(const v8bf*)(bsp + 16), CAT16);
        const __bf16* bnp = &sW[1][c * SWS + kk * 32 + klo];
        Bn[kk] = __builtin_shufflevector(*(const v8bf*)bnp,
                                         *(const v8bf*)(bnp + 16), CAT16);
      }

      v8f acc = {};                        // inline-0 C operand, no movs
#pragma unroll
      for (int kk = 0; kk < 4; ++kk) {
        acc = __builtin_amdgcn_wmma_f32_16x16x32_bf16(
            false, Ah[kk], false, Bs[kk], (short)0, acc, false, false);
        acc = __builtin_amdgcn_wmma_f32_16x16x32_bf16(
            false, An[kk], false, Bn[kk], (short)0, acc, false, false);
      }

      // Epilogue: bias + ReLU + graph norm; store pre-BN; BN statistics.
      const float bc = bv[ct];
      float psum = 0.0f, psq = 0.0f;
#pragma unroll
      for (int v = 0; v < 8; ++v) {
        const int r = rowbase + mbase + v;
        float val = fmaxf(acc[v] + bc, 0.0f) * sn[v];
        out[(size_t)r * DD + c] = val;
        psum += val;
        psq  += val * val;
      }
      atomicAdd(&ssum[c], psum);
      atomicAdd(&ssq[c],  psq);
    }
  }

  __syncthreads();
  if (t < DD) {
    atomicAdd(gsum + t, ssum[t]);
    atomicAdd(gsq  + t, ssq[t]);
  }
}

// ---------------------------------------------------------------- kernel 4
// BN finalize + residual, in-place on d_out.
__global__ __launch_bounds__(256) void bn_kernel(
    const float* __restrict__ h, const float* __restrict__ gamma,
    const float* __restrict__ beta, const float* __restrict__ gsum,
    const float* __restrict__ gsq, float* __restrict__ out) {
  const int i = blockIdx.x * blockDim.x + threadIdx.x;
  if (i >= NN * DD) return;
  const int c = i & (DD - 1);
  const float mean = gsum[c] * (1.0f / NN);
  const float var  = gsq[c] * (1.0f / NN) - mean * mean;
  const float val  = out[i];
  out[i] = h[i] + (val - mean) * rsqrtf(var + BN_EPS) * gamma[c] + beta[c];
}

// ----------------------------------------------------------------
extern "C" void kernel_launch(void* const* d_in, const int* in_sizes, int n_in,
                              void* d_out, int out_size, void* d_ws, size_t ws_size,
                              hipStream_t stream) {
  const float* h      = (const float*)d_in[0];
  const float* snorm  = (const float*)d_in[1];
  const float* Wself  = (const float*)d_in[2];
  const float* Wneigh = (const float*)d_in[3];
  const float* bias   = (const float*)d_in[4];
  const float* gamma  = (const float*)d_in[5];
  const float* beta   = (const float*)d_in[6];
  const int*   src    = (const int*)d_in[7];
  const int*   dst    = (const int*)d_in[8];
  float* out = (float*)d_out;

  float* ws   = (float*)d_ws;
  float* agg  = ws;                          // NN*DD
  float* deg  = agg + (size_t)NN * DD;       // NN
  float* gsum = deg + NN;                    // DD
  float* gsq  = gsum + DD;                   // DD

  const int zn = NN * DD + NN + 2 * DD;
  zero_kernel<<<(zn + 255) / 256, 256, 0, stream>>>(ws, zn);

  const long long ethreads = (long long)NE * 32;
  edge_kernel<<<(int)((ethreads + 255) / 256), 256, 0, stream>>>(
      h, src, dst, agg, deg);

  sage_gemm_kernel<<<(NN + 127) / 128, 256, 0, stream>>>(
      h, snorm, Wself, Wneigh, bias, agg, deg, out, gsum, gsq);

  bn_kernel<<<(NN * DD + 255) / 256, 256, 0, stream>>>(
      h, gamma, beta, gsum, gsq, out);
}
